// Embedder_86328842649751
// MI455X (gfx1250) — matrix operand order repaired and checked by reading, not compiled, not run on hardware
//
#include <hip/hip_runtime.h>
#include <cstdint>
#include <cstddef>

// ---------------------------------------------------------------------------
// Problem constants (from reference): B=128, T=256, D=128, H=512, L=3
// ---------------------------------------------------------------------------
static constexpr int Bdim = 128;
static constexpr int Tdim = 256;
static constexpr int Ddim = 128;
static constexpr int Hdim = 512;
static constexpr int H3   = 3 * Hdim;   // 1536

// CDNA5 WMMA vector types (wave32, v_wmma_f32_16x16x32_bf16)
typedef __attribute__((ext_vector_type(16))) __bf16        v16bf;
typedef __attribute__((ext_vector_type(8)))  float         v8f;

union FragU {
  unsigned short us[16];
  uint4          q[2];
  v16bf          bf;
};

__device__ __forceinline__ unsigned short f2bf(float f) {
  unsigned u = __builtin_bit_cast(unsigned, f);
  unsigned r = (u + 0x7FFFu + ((u >> 16) & 1u)) >> 16;   // round-to-nearest-even
  return (unsigned short)r;
}

__device__ __forceinline__ float sigmoidf_(float x) {
  return 1.0f / (1.0f + __expf(-x));
}

// ---------------------------------------------------------------------------
// Packed bf16 B-fragment layout (global AND the LDS mirror):
//   frag(ntile, kt) occupies 512 ushorts at (ntile*nKt + kt)*512:
//     [half(1)][lane(5)][i(3)]  ->  half*256 + lane*8 + i
// Lane mapping per ISA (16-bit B 32x16):
//   col = ntile*16 + (lane&15),  k = kt*32 + (lane>>4)*8 + half*16 + i
// Each lane's fragment = two 16-byte chunks at lane-stride 16B: global loads
// are two fully coalesced 512B wave transactions; LDS loads hit all 64 banks
// in the minimum 2 passes per ds_load_b128.
// ---------------------------------------------------------------------------
__device__ __forceinline__ v16bf load_bfrag(const unsigned short* __restrict__ packed,
                                            int ntile, int ktile, int nKt, int lane) {
  const unsigned short* base = packed + (size_t)(ntile * nKt + ktile) * 512 + lane * 8;
  FragU f;
  f.q[0] = *(const uint4*)(base);
  f.q[1] = *(const uint4*)(base + 256);
  return f.bf;
}

__device__ __forceinline__ v16bf load_bfrag_lds(const unsigned short* sm,
                                                int gate, int ktile, int lane) {
  const unsigned short* base = sm + (size_t)(gate * 16 + ktile) * 512 + lane * 8;
  FragU f;
  f.q[0] = *(const uint4*)(base);
  f.q[1] = *(const uint4*)(base + 256);
  return f.bf;
}

// ---------------------------------------------------------------------------
// A fragment directly from row-major bf16 activations — no conversion VALU.
// ISA layout (16-bit A 16x32): row = lane&15, khalf = lane>>4,
//   i<8  -> K = ktile*32 + khalf*8 + i ;  i>=8 -> +16
// => two contiguous 16-byte chunks per lane (2x global_load_b128).
// ---------------------------------------------------------------------------
__device__ __forceinline__ v16bf load_afrag_bf16(const unsigned short* __restrict__ A,
                                                 int lda, int mtile, int ktile, int lane) {
  const int row   = mtile * 16 + (lane & 15);
  const int khalf = lane >> 4;
  const unsigned short* base = A + (size_t)row * lda + ktile * 32 + khalf * 8;
  FragU f;
  f.q[0] = *(const uint4*)(base);
  f.q[1] = *(const uint4*)(base + 16);
  return f.bf;
}

// ---------------------------------------------------------------------------
// Device-scope grid barrier (persistent-kernel step sync).
// ---------------------------------------------------------------------------
__device__ __forceinline__ void grid_sync(unsigned* cnt, int phase) {
  __syncthreads();
  if (threadIdx.x == 0) {
    __threadfence();                     // release all stores device-wide
    atomicAdd(cnt, 1u);
    const unsigned target = (unsigned)gridDim.x * (unsigned)(phase + 1);
    while (__hip_atomic_load(cnt, __ATOMIC_ACQUIRE, __HIP_MEMORY_SCOPE_AGENT) < target) {
      __builtin_amdgcn_s_sleep(2);
    }
  }
  __syncthreads();
}

__global__ void k_zero_counters(unsigned* __restrict__ cnt, int n) {
  int i = threadIdx.x;
  if (i < n) cnt[i] = 0u;
}

// ---------------------------------------------------------------------------
// Kernel: transpose + convert X [B,T,D] f32 -> Xt [T,B,D] bf16
// ---------------------------------------------------------------------------
__global__ __launch_bounds__(256) void k_transpose_x(const float* __restrict__ X,
                                                     unsigned short* __restrict__ Xt) {
  size_t gid = (size_t)blockIdx.x * blockDim.x + threadIdx.x;
  const size_t total = (size_t)Tdim * Bdim * Ddim;
  if (gid >= total) return;
  int t = (int)(gid / ((size_t)Bdim * Ddim));
  int r = (int)(gid % ((size_t)Bdim * Ddim));
  int b = r / Ddim;
  int d = r % Ddim;
  Xt[gid] = f2bf(X[((size_t)b * Tdim + t) * Ddim + d]);
}

// ---------------------------------------------------------------------------
// Kernel: pack f32 [K,N] weights into the half-split bf16 fragment layout.
//   gid = ((ntile*nKt + kt)*2 + half)*256 + lane*8 + i
// ---------------------------------------------------------------------------
__global__ __launch_bounds__(256) void k_pack_b(const float* __restrict__ Bm,
                                                unsigned short* __restrict__ out,
                                                int K, int N) {
  size_t gid = (size_t)blockIdx.x * blockDim.x + threadIdx.x;
  const size_t total = (size_t)K * N;
  if (gid >= total) return;
  const int nKt = K >> 5;
  int i    = (int)(gid & 7);
  int lane = (int)((gid >> 3) & 31);
  int half = (int)((gid >> 8) & 1);
  size_t frag = gid >> 9;
  int kt    = (int)(frag % nKt);
  int ntile = (int)(frag / nKt);
  int col = ntile * 16 + (lane & 15);
  int k   = kt * 32 + (lane >> 4) * 8 + half * 16 + i;
  out[gid] = f2bf(Bm[(size_t)k * N + col]);
}

// ---------------------------------------------------------------------------
// Kernel: bf16 WMMA GEMM, C[M,N] = act(Abf16[M,K] @ Bpacked + bias[N]).
// Wave tile 16x64 (1 A frag reused across 4 B frags / K-step), 8 waves/block.
// act: 0 none, 1 sigmoid. trans_out: row m = t*B+b -> store [(b*T+t), n].
// ---------------------------------------------------------------------------
__global__ __launch_bounds__(256) void k_wmma_gemm_bias(const unsigned short* __restrict__ A,
                                                        const unsigned short* __restrict__ Bp,
                                                        const float* __restrict__ bias,
                                                        float* __restrict__ Cf,
                                                        int M, int K, int N,
                                                        int act, int trans_out) {
  const int lane   = threadIdx.x & 31;
  const int wave   = threadIdx.x >> 5;
  const int Mtiles = M >> 4;
  const int wg     = blockIdx.x * 8 + wave;
  const int mtile  = wg % Mtiles;
  const int ng     = wg / Mtiles;          // 64-col group
  const int nKt    = K >> 5;

  v8f acc[4] = {};

  for (int kt = 0; kt < nKt; ++kt) {
    v16bf a = load_afrag_bf16(A, K, mtile, kt, lane);
#pragma unroll
    for (int j = 0; j < 4; ++j) {
      v16bf b = load_bfrag(Bp, ng * 4 + j, kt, nKt, lane);
      acc[j] = __builtin_amdgcn_wmma_f32_16x16x32_bf16(
          false, a, false, b, (short)0, acc[j], false, false);
    }
  }

  const int khalf = lane >> 4;
  const int nlo   = lane & 15;
#pragma unroll
  for (int g = 0; g < 8; ++g) {
    int row = mtile * 16 + khalf * 8 + g;
#pragma unroll
    for (int j = 0; j < 4; ++j) {
      int n = ng * 64 + j * 16 + nlo;
      float v = acc[j][g] + bias[n];
      if (act) v = sigmoidf_(v);
      size_t idx;
      if (trans_out) {
        int t = row / Bdim;
        int b = row % Bdim;
        idx = ((size_t)b * Tdim + t) * N + n;
      } else {
        idx = (size_t)row * N + n;
      }
      Cf[idx] = v;
    }
  }
}

// ---------------------------------------------------------------------------
// Kernel: persistent GRU layer scan. One launch runs all T=256 timesteps,
// separated by a device-scope grid barrier. 32 blocks x 256 threads.
// Remap vs round 2: ct = blockIdx (H-col tile), mtile = waveId (batch tile)
// so all 8 waves of a block share ONE 48KB U-fragment set -> staged in LDS
// once, then served from ds_load_b128 for all 256 steps (cuts per-layer L2
// weight traffic from ~384MB to 1.5MB on the sequential critical path).
//   gh = h_{t-1} @ U + b_rec ;  z=sig(xz+hz); r=sig(xr+hr);
//   hc = tanh(xh + r*hh);  h_t = z*h_{t-1} + (1-z)*hc
// ---------------------------------------------------------------------------
__global__ __launch_bounds__(256) void k_gru_layer(const float* __restrict__ gx,      // [T,B,3H] f32
                                                   const unsigned short* __restrict__ Up,
                                                   const float* __restrict__ brec,    // [3H]
                                                   unsigned short* __restrict__ xout, // [T,B,H] bf16
                                                   float* __restrict__ hping,         // [2,B,H] f32
                                                   unsigned* __restrict__ barrier_cnt) {
  const int lane  = threadIdx.x & 31;
  const int wave  = threadIdx.x >> 5;
  const int ct    = blockIdx.x;  // 0..31  (H col tile, shared by the block)
  const int mtile = wave;        // 0..7   (batch tile, per wave)
  const int nKt   = Hdim >> 5;   // 16
  const int khalf = lane >> 4;
  const int nlo   = lane & 15;
  const int n     = ct * 16 + nlo;

  // ---- stage this block's 3 gate weight tiles into LDS (48 KB) ------------
  __shared__ __align__(16) unsigned short smem[3 * 16 * 512];   // 49152 B
  {
    uint4*       dst = (uint4*)smem;                 // 3072 chunks of 16B
    const uint4* src = (const uint4*)Up;
    for (int c = threadIdx.x; c < 3072; c += 256) {
      int g = c >> 10;                               // gate
      int r = c & 1023;
      dst[g * 1024 + r] = src[(size_t)(ct + g * 32) * 1024 + r];
    }
  }
  __syncthreads();

  const float bz = brec[n];
  const float br = brec[n + Hdim];
  const float bh = brec[n + 2 * Hdim];

  for (int t = 0; t < Tdim; ++t) {
    v8f accz = {}, accr = {}, acch = {};

    if (t > 0) {
      const unsigned short* hbf = xout + (size_t)(t - 1) * Bdim * Hdim;
      for (int kt = 0; kt < nKt; ++kt) {
        v16bf a  = load_afrag_bf16(hbf, Hdim, mtile, kt, lane);
        v16bf vz = load_bfrag_lds(smem, 0, kt, lane);
        v16bf vr = load_bfrag_lds(smem, 1, kt, lane);
        v16bf vh = load_bfrag_lds(smem, 2, kt, lane);
        accz = __builtin_amdgcn_wmma_f32_16x16x32_bf16(false, a, false, vz, (short)0, accz, false, false);
        accr = __builtin_amdgcn_wmma_f32_16x16x32_bf16(false, a, false, vr, (short)0, accr, false, false);
        acch = __builtin_amdgcn_wmma_f32_16x16x32_bf16(false, a, false, vh, (short)0, acch, false, false);
      }
    }

    const float* gxt   = gx + (size_t)t * Bdim * H3;
    const float* hprev = hping + (size_t)((t + 1) & 1) * Bdim * Hdim;  // slot (t-1)&1
    float*       hcur  = hping + (size_t)(t & 1) * Bdim * Hdim;
    unsigned short* xo = xout + (size_t)t * Bdim * Hdim;
    const float* gxnext = gx + (size_t)(t + 1) * Bdim * H3;

#pragma unroll
    for (int g = 0; g < 8; ++g) {
      int b = mtile * 16 + khalf * 8 + g;             // batch row
      const float* gxr = gxt + (size_t)b * H3;
      float xz = gxr[n];
      float xr = gxr[n + Hdim];
      float xh = gxr[n + 2 * Hdim];
      float hp = (t > 0) ? hprev[(size_t)b * Hdim + n] : 0.0f;
      float z  = sigmoidf_(xz + accz[g] + bz);
      float r  = sigmoidf_(xr + accr[g] + br);
      float hc = tanhf(xh + r * (acch[g] + bh));
      float hn = z * hp + (1.0f - z) * hc;
      hcur[(size_t)b * Hdim + n] = hn;        // f32 state for next step's gates
      xo[(size_t)b * Hdim + n]   = f2bf(hn);  // bf16 for next matmuls

      if (t + 1 < Tdim) {                     // hide next step's gx behind WMMAs
        const float* gxp = gxnext + (size_t)b * H3;
        __builtin_prefetch(gxp + n, 0, 1);            // -> global_prefetch_b8
        __builtin_prefetch(gxp + n + Hdim, 0, 1);
        __builtin_prefetch(gxp + n + 2 * Hdim, 0, 1);
      }
    }

    grid_sync(barrier_cnt, t);
  }
}

// ---------------------------------------------------------------------------
// Host driver
// ---------------------------------------------------------------------------
extern "C" void kernel_launch(void* const* d_in, const int* in_sizes, int n_in,
                              void* d_out, int out_size, void* d_ws, size_t ws_size,
                              hipStream_t stream) {
  const float* X            = (const float*)d_in[0];   // [B,T,D]
  const float* kernel0      = (const float*)d_in[1];   // [D,3H]
  const float* kernels_rest = (const float*)d_in[2];   // [L-1,H,3H]
  const float* rec_kernels  = (const float*)d_in[3];   // [L,H,3H]
  const float* biases       = (const float*)d_in[4];   // [L,2,3H]
  const float* Wd           = (const float*)d_in[5];   // [H,H]
  const float* bd           = (const float*)d_in[6];   // [H]
  float* out                = (float*)d_out;           // [B,T,H] f32

  // ---- workspace layout (256-byte aligned) --------------------------------
  char* ws = (char*)d_ws;
  size_t off = 0;
  auto alloc = [&](size_t bytes) -> char* {
    char* p = ws + off;
    off = (off + bytes + 255) & ~(size_t)255;
    return p;
  };
  const size_t M = (size_t)Tdim * Bdim;   // 32768 rows

  unsigned short* Xt   = (unsigned short*)alloc(M * Ddim * 2);   //  8 MB  bf16
  unsigned short* xbf0 = (unsigned short*)alloc(M * Hdim * 2);   // 32 MB  bf16
  unsigned short* xbf1 = (unsigned short*)alloc(M * Hdim * 2);   // 32 MB  bf16
  float* gx            = (float*)alloc(M * H3 * sizeof(float));  // 192 MB f32
  float* hping         = (float*)alloc((size_t)2 * Bdim * Hdim * sizeof(float));
  unsigned short* pk0  = (unsigned short*)alloc((size_t)Ddim * H3 * 2);
  unsigned short* pk1  = (unsigned short*)alloc((size_t)Hdim * H3 * 2);
  unsigned short* pk2  = (unsigned short*)alloc((size_t)Hdim * H3 * 2);
  unsigned short* pu0  = (unsigned short*)alloc((size_t)Hdim * H3 * 2);
  unsigned short* pu1  = (unsigned short*)alloc((size_t)Hdim * H3 * 2);
  unsigned short* pu2  = (unsigned short*)alloc((size_t)Hdim * H3 * 2);
  unsigned short* pwd  = (unsigned short*)alloc((size_t)Hdim * Hdim * 2);
  unsigned* counters   = (unsigned*)alloc(64);                   // 3 grid barriers

  // ---- 0) reset barrier counters ------------------------------------------
  k_zero_counters<<<dim3(1), dim3(64), 0, stream>>>(counters, 3);

  // ---- 1) transpose+convert X -> [T,B,D] bf16 -----------------------------
  {
    size_t total = M * Ddim;
    k_transpose_x<<<dim3((unsigned)((total + 255) / 256)), dim3(256), 0, stream>>>(X, Xt);
  }

  // ---- 2) pack all weights to the half-split bf16 fragment layout ---------
  auto pack = [&](const float* src, unsigned short* dst, int K, int N) {
    size_t total = (size_t)K * N;
    k_pack_b<<<dim3((unsigned)((total + 255) / 256)), dim3(256), 0, stream>>>(src, dst, K, N);
  };
  pack(kernel0,                              pk0, Ddim, H3);
  pack(kernels_rest + 0 * (size_t)Hdim * H3, pk1, Hdim, H3);
  pack(kernels_rest + 1 * (size_t)Hdim * H3, pk2, Hdim, H3);
  pack(rec_kernels  + 0 * (size_t)Hdim * H3, pu0, Hdim, H3);
  pack(rec_kernels  + 1 * (size_t)Hdim * H3, pu1, Hdim, H3);
  pack(rec_kernels  + 2 * (size_t)Hdim * H3, pu2, Hdim, H3);
  pack(Wd,                                   pwd, Hdim, Hdim);

  // ---- 3) three GRU layers ------------------------------------------------
  struct Layer {
    const unsigned short* in; int K;
    const unsigned short* Wk; const unsigned short* Uk;
    const float* bias2; unsigned short* outb;
  };
  Layer layers[3] = {
    { Xt,   Ddim, pk0, pu0, biases + 0 * 2 * H3, xbf0 },
    { xbf0, Hdim, pk1, pu1, biases + 1 * 2 * H3, xbf1 },
    { xbf1, Hdim, pk2, pu2, biases + 2 * 2 * H3, xbf0 },
  };

  for (int l = 0; l < 3; ++l) {
    const Layer& L = layers[l];
    // gx_all = x @ Wk + b_in for all T*B rows (time-parallel WMMA GEMM)
    {
      unsigned blocks = (unsigned)(((M / 16) * (H3 / 64)) / 8);   // 6144
      k_wmma_gemm_bias<<<dim3(blocks), dim3(256), 0, stream>>>(
          L.in, L.Wk, L.bias2 /*b_in*/, gx, (int)M, L.K, H3, /*act=*/0, /*trans_out=*/0);
    }
    // persistent sequential scan over all T timesteps (one launch)
    k_gru_layer<<<dim3(32), dim3(256), 0, stream>>>(
        gx, L.Uk, L.bias2 + H3 /*b_rec*/, L.outb, hping, counters + l);
  }

  // ---- 4) dense head: sigmoid(x3 @ Wd + bd), stored transposed [B,T,H] ----
  {
    unsigned blocks = (unsigned)(((M / 16) * (Hdim / 64)) / 8);   // 2048
    k_wmma_gemm_bias<<<dim3(blocks), dim3(256), 0, stream>>>(
        xbf0, pwd, bd, out, (int)M, Hdim, Hdim, /*act=*/1, /*trans_out=*/1);
  }
}
